// CoxPHLoss_43044162241128
// MI455X (gfx1250) — compile-verified, matrix operand order checked by reading, not run on hardware
//
#include <hip/hip_runtime.h>
#include <hip/hip_bf16.h>
#include <math.h>

typedef __attribute__((ext_vector_type(16))) _Float16 v16h;
typedef __attribute__((ext_vector_type(8)))  _Float16 v8h;
typedef __attribute__((ext_vector_type(4)))  _Float16 v4h;
typedef __attribute__((ext_vector_type(8)))  float    v8f;
typedef __attribute__((ext_vector_type(4)))  float    v4f;

#define CHUNK 1024   // j's staged in LDS per block iteration (divides 16384)

// ---------------------------------------------------------------------------
// acc[0] = loss numerator  Σ (theta_i - log S_i) * event_i
// acc[1] = event sum       Σ event_i
// ---------------------------------------------------------------------------
__global__ void cox_init_kernel(float* __restrict__ acc) {
    acc[0] = 0.0f;
    acc[1] = 0.0f;
}

// exp(risk) -> f16 table, and wave-reduced event sum.
__global__ void cox_prep_kernel(const float* __restrict__ risk,
                                const float* __restrict__ event,
                                _Float16* __restrict__ ef,
                                float* __restrict__ acc,
                                int n) {
    int i = blockIdx.x * blockDim.x + threadIdx.x;
    float e = 0.0f;
    if (i < n) {
        ef[i] = (_Float16)expf(risk[i]);
        e = event[i];
    }
    #pragma unroll
    for (int off = 16; off >= 1; off >>= 1)
        e += __shfl_down(e, off, 32);
    if ((threadIdx.x & 31) == 0 && e != 0.0f)
        atomicAdd(&acc[1], e);
}

// ---------------------------------------------------------------------------
// Main O(N^2) sweep. One wave owns a 16-row tile; the block stages CHUNK j's
// of (time f32, exp f16) in LDS once, then each wave runs CHUNK/32 iterations
// of V_WMMA_F32_16X16X32_F16 with an all-ones B so every D column holds the
// masked row sum (f32 accumulation).
//
// A-tile layout (ISA 7.12.2, 16-bit A 16x32):
//   lane L: row m = L%16, base = 8*(L/16)
//   half slots h=0..7  -> K = base + h        (VGPR0..3)
//   half slots h=8..15 -> K = base + 16 + h-8 (VGPR4..7)
// D layout: lane L reg r -> row = r + 8*(L/16), col = L%16 (cols identical).
// LDS reads are half-wave broadcasts (same address within lanes 0-15 and
// within 16-31) -> conflict-free.
// ---------------------------------------------------------------------------
__global__ void __launch_bounds__(256)
cox_wmma_kernel(const float* __restrict__ risk,
                const float* __restrict__ tvals,
                const float* __restrict__ event,
                const _Float16* __restrict__ ef,
                float* __restrict__ acc,
                int n) {
    __shared__ __align__(16) float    s_time[CHUNK];
    __shared__ __align__(16) _Float16 s_ef[CHUNK];

    const int tid   = threadIdx.x;
    const int lane  = tid & 31;
    const int wave  = tid >> 5;
    const int tile  = blockIdx.x * 8 + wave;   // 16 rows per tile
    const int ibase = tile * 16;
    const int m     = lane & 15;
    const int kb    = (lane >> 4) << 3;        // 0 or 8

    const float t_row = tvals[ibase + m];

    v16h b;
    #pragma unroll
    for (int h = 0; h < 16; ++h) b[h] = (_Float16)1.0f;

    v8f c = {};

    for (int jb = 0; jb < n; jb += CHUNK) {
        // ---- cooperative staging: 4KB time + 2KB ef per chunk ----
        ((v4f*)s_time)[tid] = ((const v4f*)(tvals + jb))[tid];   // 16B/thread
        ((v4h*)s_ef)[tid]   = ((const v4h*)(ef + jb))[tid];      // 8B/thread
        // prefetch next chunk while this one is consumed
        if (jb + CHUNK < n) {
            __builtin_prefetch((const void*)(tvals + jb + CHUNK), 0, 0);
            __builtin_prefetch((const void*)(ef + jb + CHUNK), 0, 0);
        }
        __syncthreads();

        #pragma unroll 4
        for (int jj = 0; jj < CHUNK; jj += 32) {
            const int j0 = jj + kb;        // 8 contiguous halves / floats
            const int j1 = jj + kb + 16;

            v8h e0 = *(const v8h*)(s_ef + j0);      // ds_load_b128
            v8h e1 = *(const v8h*)(s_ef + j1);
            v8f t0 = *(const v8f*)(s_time + j0);    // 2x ds_load_b128
            v8f t1 = *(const v8f*)(s_time + j1);

            v16h a;
            #pragma unroll
            for (int h = 0; h < 8; ++h)
                a[h] = (t0[h] >= t_row) ? e0[h] : (_Float16)0.0f;
            #pragma unroll
            for (int h = 0; h < 8; ++h)
                a[8 + h] = (t1[h] >= t_row) ? e1[h] : (_Float16)0.0f;

            // D = A x ones + C : every column of D = masked row sum (f32)
            c = __builtin_amdgcn_wmma_f32_16x16x32_f16(
                    /*neg_a=*/false, a, /*neg_b=*/false, b,
                    /*c_mod=*/(short)0, c, /*reuse_a=*/false, /*reuse_b=*/false);
        }
        __syncthreads();
    }

    // Column 0 lives in lanes 0 (rows 0..7) and 16 (rows 8..15).
    if ((lane & 15) == 0) {
        const int rbase = ibase + ((lane >> 4) << 3);
        float local = 0.0f;
        #pragma unroll
        for (int r = 0; r < 8; ++r) {
            const int g = rbase + r;
            const float S = c[r];              // S >= exp(risk[g]) > 0
            local += (risk[g] - logf(S)) * event[g];
        }
        atomicAdd(&acc[0], local);
    }
}

__global__ void cox_final_kernel(const float* __restrict__ acc,
                                 float* __restrict__ out) {
    out[0] = -acc[0] / acc[1];
}

// ---------------------------------------------------------------------------
extern "C" void kernel_launch(void* const* d_in, const int* in_sizes, int n_in,
                              void* d_out, int out_size, void* d_ws, size_t ws_size,
                              hipStream_t stream) {
    const float* risk  = (const float*)d_in[0];
    const float* tvals = (const float*)d_in[1];
    const float* event = (const float*)d_in[2];
    float* out = (float*)d_out;
    const int n = in_sizes[0];   // 16384 (multiple of CHUNK by construction)

    // workspace layout: [0, 2n) bytes = f16 exp table ; then 2 floats of accum
    _Float16* ef = (_Float16*)d_ws;
    float* acc = (float*)((char*)d_ws + (size_t)n * sizeof(_Float16));

    cox_init_kernel<<<1, 1, 0, stream>>>(acc);

    {
        const int threads = 256;
        const int blocks = (n + threads - 1) / threads;
        cox_prep_kernel<<<blocks, threads, 0, stream>>>(risk, event, ef, acc, n);
    }

    {
        const int tiles = n / 16;          // 1024 row tiles
        const int blocks = tiles / 8;      // 8 waves (tiles) per 256-thread block
        cox_wmma_kernel<<<blocks, 256, 0, stream>>>(risk, tvals, event, ef, acc, n);
    }

    cox_final_kernel<<<1, 1, 0, stream>>>(acc, out);
}
